// MambaEncoding_41068477285023
// MI455X (gfx1250) — matrix operand order repaired and checked by reading, not compiled, not run on hardware
//
#include <hip/hip_runtime.h>
#include <hip/hip_bf16.h>
#include <cstdint>

// Problem constants (from reference)
#define STIM    1024
#define HID     256
#define VOX     8192
#define D_INNER 512
#define D_STATE 16
#define D_CONV  4
#define DT_RANK 16
#define NB      8
#define LSEQ    2048
#define MROWS   (NB * LSEQ)   // 16384
#define NXDB    (DT_RANK + 2 * D_STATE)  // 48

typedef __attribute__((ext_vector_type(16))) __bf16 v16bf;
typedef __attribute__((ext_vector_type(8)))  float  v8f;
typedef __attribute__((ext_vector_type(4)))  int    i4;

#if defined(__gfx1250__) && __has_builtin(__builtin_amdgcn_global_load_async_to_lds_b128)
#define HAVE_ASYNC 1
#else
#define HAVE_ASYNC 0
#endif

__device__ __forceinline__ unsigned short f2bf(float f) {
    union { float f; unsigned int u; } c; c.f = f;
    unsigned int x = c.u;
    x += 0x7FFFu + ((x >> 16) & 1u);   // round-to-nearest-even
    return (unsigned short)(x >> 16);
}

#if HAVE_ASYNC
typedef __attribute__((address_space(1))) i4 gi4;   // global (__device__) int4
typedef __attribute__((address_space(3))) i4 li4;   // LDS (__shared__) int4
#endif

// 16-byte global->LDS copy: async (ASYNCcnt) if available, else sync via VGPRs.
__device__ __forceinline__ void cp16_g2l(const unsigned short* g, unsigned short* l) {
#if HAVE_ASYNC
    __builtin_amdgcn_global_load_async_to_lds_b128(
        (gi4*)(uintptr_t)g, (li4*)l, 0, 0);
#else
    *(uint4*)l = *(const uint4*)g;
#endif
}
__device__ __forceinline__ void wait_async_le3() {
#if HAVE_ASYNC
    asm volatile("s_wait_asynccnt 0x3" ::: "memory");
#endif
}
__device__ __forceinline__ void wait_async_0() {
#if HAVE_ASYNC
    asm volatile("s_wait_asynccnt 0x0" ::: "memory");
#endif
}

// ---------------------------------------------------------------------------
// GEMM: C[M,N] = A[M,K](bf16,row-major) * W[N,K](bf16,row-major)^T (+bias[N])
// C stored as f32 (c_bf16=0) or bf16 (c_bf16=1).
// Block 256 threads (8 waves). Tile 128(M) x 64(N), K step 32, double-buffered
// LDS fed by GLOBAL_LOAD_ASYNC_TO_LDS_B128. Wave w owns M sub-tile w and all
// four N sub-tiles -> 4 x v_wmma_f32_16x16x32_bf16 per k-step (issued
// back-to-back after one batched fragment-load group).
// Requires: M % 128 == 0, K % 32 == 0. N guarded (pad columns clamped).
// ---------------------------------------------------------------------------
#define TM 128
#define TN 64
#define TK 32

__global__ __launch_bounds__(256)
void gemm_bf16_wmma(const unsigned short* __restrict__ A,
                    const unsigned short* __restrict__ W,
                    const float* __restrict__ bias,
                    void* __restrict__ Cout, int c_bf16,
                    int M, int N, int K) {
    __shared__ __align__(16) unsigned short lA[2][TM * TK];  // 8 KB x2
    __shared__ __align__(16) unsigned short lB[2][TN * TK];  // 4 KB x2

    const int tid  = threadIdx.x;
    const int lane = tid & 31;
    const int wave = tid >> 5;          // 0..7 = M sub-tile
    const int half = lane >> 4;
    const int l15  = lane & 15;
    const int m0   = blockIdx.y * TM;
    const int n0   = blockIdx.x * TN;

    // Staging map: 16B chunk = 8 bf16. A tile: 512 chunks (2/thread),
    // B tile: 256 chunks (1/thread). chunk q -> row q>>2, col (q&3)*8.
    const int ar0 = tid >> 2;
    const int ac  = (tid & 3) * 8;
    const int ar1 = ar0 + 64;
    const int br  = tid >> 2;
    const int bnc = min(n0 + br, N - 1);  // clamp: pad cols never stored

    const unsigned short* gA0 = A + (size_t)(m0 + ar0) * K + ac;
    const unsigned short* gA1 = A + (size_t)(m0 + ar1) * K + ac;
    const unsigned short* gB  = W + (size_t)bnc * K + ac;

    v8f acc[4];
    #pragma unroll
    for (int j = 0; j < 4; ++j) acc[j] = v8f{};

    // prologue: stage k0=0 into buffer 0
    cp16_g2l(gA0, &lA[0][ar0 * TK + ac]);
    cp16_g2l(gA1, &lA[0][ar1 * TK + ac]);
    cp16_g2l(gB,  &lB[0][br  * TK + ac]);

    int buf = 0;
    for (int k0 = 0; k0 < K; k0 += TK) {
        const int k1 = k0 + TK;
        if (k1 < K) {
            // stage next tile into the other buffer (safe: last reads of that
            // buffer were fenced by the barrier at the end of the previous step)
            cp16_g2l(gA0 + k1, &lA[buf ^ 1][ar0 * TK + ac]);
            cp16_g2l(gA1 + k1, &lA[buf ^ 1][ar1 * TK + ac]);
            cp16_g2l(gB  + k1, &lB[buf ^ 1][br  * TK + ac]);
            wait_async_le3();    // current tile's 3 async ops complete
        } else {
            wait_async_0();
        }
        __syncthreads();

        // fragments: A 16x32 (lane=row M; e<8 -> K=8*half+e, e>=8 -> 16+8*half+e-8)
        //            B 32x16 (lane=col N; same K order as A)
        union FragU { uint4 u4[2]; v16bf v; };
        FragU fa, fb[4];
        {
            const unsigned short* p = &lA[buf][(wave * 16 + l15) * TK + half * 8];
            fa.u4[0] = *(const uint4*)p;
            fa.u4[1] = *(const uint4*)(p + 16);
        }
        #pragma unroll
        for (int j = 0; j < 4; ++j) {
            const unsigned short* p = &lB[buf][(j * 16 + l15) * TK + half * 8];
            fb[j].u4[0] = *(const uint4*)p;
            fb[j].u4[1] = *(const uint4*)(p + 16);
        }
        // all LDS loads issued; WMMAs can now go back-to-back
        #pragma unroll
        for (int j = 0; j < 4; ++j)
            acc[j] = __builtin_amdgcn_wmma_f32_16x16x32_bf16(
                         false, fa.v, false, fb[j].v, (short)0, acc[j], false, false);
        __syncthreads();
        buf ^= 1;
    }

    // epilogue: C/D layout: VGPR i -> M = i + 8*half, lane l15 -> N
    #pragma unroll
    for (int j = 0; j < 4; ++j) {
        const int n = n0 + j * 16 + l15;
        if (n < N) {
            const float bv = bias ? bias[n] : 0.f;
            #pragma unroll
            for (int i = 0; i < 8; ++i) {
                const int m = m0 + wave * 16 + half * 8 + i;
                const float v = acc[j][i] + bv;
                if (c_bf16) ((unsigned short*)Cout)[(size_t)m * N + n] = f2bf(v);
                else        ((float*)Cout)[(size_t)m * N + n] = v;
            }
        }
    }
}

// ---------------------------------------------------------------------------
// f32 -> bf16 bulk convert (8 elems/thread, b128 in / b64 out)
// ---------------------------------------------------------------------------
__global__ __launch_bounds__(256)
void f32_to_bf16_kernel(const float* __restrict__ src,
                        unsigned short* __restrict__ dst, int n8) {
    int i = blockIdx.x * 256 + threadIdx.x;
    if (i >= n8) return;
    const float4* s = (const float4*)src + (size_t)i * 2;
    float4 a = s[0], b = s[1];
    ushort4 r0 = {f2bf(a.x), f2bf(a.y), f2bf(a.z), f2bf(a.w)};
    ushort4 r1 = {f2bf(b.x), f2bf(b.y), f2bf(b.z), f2bf(b.w)};
    ushort4* d = (ushort4*)dst + (size_t)i * 2;
    d[0] = r0; d[1] = r1;
}

// ---------------------------------------------------------------------------
// Depthwise causal conv (D_CONV=4) + SiLU. Emits f32 (for scan) + bf16 (GEMM).
// ---------------------------------------------------------------------------
__global__ __launch_bounds__(256)
void conv_silu_kernel(const float* __restrict__ xz, const float* __restrict__ conv_w,
                      const float* __restrict__ conv_b,
                      float* __restrict__ uc, unsigned short* __restrict__ uc_bf) {
    int idx = blockIdx.x * 256 + threadIdx.x;     // b*L*512 + l*512 + d
    int d = idx & (D_INNER - 1);
    int l = (idx >> 9) & (LSEQ - 1);
    int b = idx >> 20;
    float acc = conv_b[d];
    #pragma unroll
    for (int k = 0; k < D_CONV; ++k) {
        int ls = l - (D_CONV - 1) + k;
        if (ls >= 0)
            acc += conv_w[d * D_CONV + k] *
                   xz[(size_t)(b * LSEQ + ls) * (2 * D_INNER) + d];
    }
    float v = acc / (1.f + __expf(-acc));
    uc[(size_t)idx] = v;
    uc_bf[(size_t)idx] = f2bf(v);
}

// ---------------------------------------------------------------------------
// dt = softplus(xdb[:, :16] @ dt_proj_w.T + dt_proj_b)
// ---------------------------------------------------------------------------
__global__ __launch_bounds__(256)
void dt_softplus_kernel(const float* __restrict__ xdb, const float* __restrict__ dt_w,
                        const float* __restrict__ dt_b, float* __restrict__ dtf) {
    int idx = blockIdx.x * 256 + threadIdx.x;
    int d   = idx & (D_INNER - 1);
    int row = idx >> 9;                           // b*L + l
    float s = dt_b[d];
    const float* xr = xdb + (size_t)row * NXDB;
    #pragma unroll
    for (int r = 0; r < DT_RANK; ++r) s += xr[r] * dt_w[d * DT_RANK + r];
    dtf[(size_t)idx] = (s > 20.f) ? s : __logf(1.f + __expf(s));
}

// ---------------------------------------------------------------------------
// Selective scan + D skip + SiLU(z) gate. One block per batch, thread = d.
// Output bf16 (feeds out_proj GEMM).
// ---------------------------------------------------------------------------
__global__ __launch_bounds__(512)
void scan_kernel(const float* __restrict__ dtf, const float* __restrict__ xdb,
                 const float* __restrict__ uc,  const float* __restrict__ xz,
                 const float* __restrict__ A_log, const float* __restrict__ Dp,
                 unsigned short* __restrict__ yg) {
    __shared__ float sB[D_STATE];
    __shared__ float sC[D_STATE];
    const int d = threadIdx.x;
    const int b = blockIdx.x;

    float a[D_STATE], st[D_STATE];
    #pragma unroll
    for (int s = 0; s < D_STATE; ++s) {
        a[s]  = -__expf(A_log[d * D_STATE + s]);
        st[s] = 0.f;
    }
    const float Dd = Dp[d];

    for (int l = 0; l < LSEQ; ++l) {
        size_t row = (size_t)b * LSEQ + l;
        if (threadIdx.x < 16)
            sB[threadIdx.x] = xdb[row * NXDB + DT_RANK + threadIdx.x];
        else if (threadIdx.x < 32)
            sC[threadIdx.x - 16] = xdb[row * NXDB + DT_RANK + D_STATE + (threadIdx.x - 16)];
        __syncthreads();

        float dt_t = dtf[row * D_INNER + d];
        float u_t  = uc [row * D_INNER + d];
        float z_t  = xz [row * (2 * D_INNER) + D_INNER + d];
        float du   = dt_t * u_t;
        float y = 0.f;
        #pragma unroll
        for (int s = 0; s < D_STATE; ++s) {
            float dA = __expf(dt_t * a[s]);
            st[s] = dA * st[s] + du * sB[s];
            y += st[s] * sC[s];
        }
        float g = z_t / (1.f + __expf(-z_t));
        yg[row * D_INNER + d] = f2bf((y + u_t * Dd) * g);
        __syncthreads();
    }
}

// ---------------------------------------------------------------------------
// Launch
// ---------------------------------------------------------------------------
extern "C" void kernel_launch(void* const* d_in, const int* in_sizes, int n_in,
                              void* d_out, int out_size, void* d_ws, size_t ws_size,
                              hipStream_t stream) {
    const float* x         = (const float*)d_in[0];
    const float* proj_w    = (const float*)d_in[1];
    const float* proj_b    = (const float*)d_in[2];
    const float* in_proj_w = (const float*)d_in[3];
    const float* conv_w    = (const float*)d_in[4];
    const float* conv_b    = (const float*)d_in[5];
    const float* x_proj_w  = (const float*)d_in[6];
    const float* dt_proj_w = (const float*)d_in[7];
    const float* dt_proj_b = (const float*)d_in[8];
    const float* A_log     = (const float*)d_in[9];
    const float* Dp        = (const float*)d_in[10];
    const float* out_proj_w= (const float*)d_in[11];
    const float* head_w    = (const float*)d_in[12];
    const float* head_b    = (const float*)d_in[13];
    float* out = (float*)d_out;
    char*  ws  = (char*)d_ws;

    size_t off = 0;
    auto alloc = [&](size_t bytes) -> void* {
        void* p = ws + off;
        off = (off + bytes + 255) & ~(size_t)255;
        return p;
    };
    typedef unsigned short bf16_t;
    bf16_t* x_bf   = (bf16_t*)alloc((size_t)MROWS * STIM * 2);
    bf16_t* pw_bf  = (bf16_t*)alloc((size_t)HID * STIM * 2);
    bf16_t* ipw_bf = (bf16_t*)alloc((size_t)2 * D_INNER * HID * 2);
    bf16_t* xpw_bf = (bf16_t*)alloc((size_t)NXDB * D_INNER * 2);
    bf16_t* opw_bf = (bf16_t*)alloc((size_t)HID * D_INNER * 2);
    bf16_t* hw_bf  = (bf16_t*)alloc((size_t)VOX * HID * 2);
    bf16_t* h_bf   = (bf16_t*)alloc((size_t)MROWS * HID * 2);
    float*  xz     = (float*) alloc((size_t)MROWS * 2 * D_INNER * 4);
    float*  uc     = (float*) alloc((size_t)MROWS * D_INNER * 4);
    bf16_t* uc_bf  = (bf16_t*)alloc((size_t)MROWS * D_INNER * 2);
    float*  xdb    = (float*) alloc((size_t)MROWS * NXDB * 4);
    float*  dtf    = (float*) alloc((size_t)MROWS * D_INNER * 4);
    bf16_t* yg_bf  = (bf16_t*)alloc((size_t)MROWS * D_INNER * 2);
    bf16_t* o_bf   = (bf16_t*)alloc((size_t)MROWS * HID * 2);

    auto cvt = [&](const float* s, bf16_t* d, size_t n) {
        int n8 = (int)(n / 8);
        f32_to_bf16_kernel<<<(n8 + 255) / 256, 256, 0, stream>>>(s, d, n8);
    };
    // one-time operand conversions (f32 -> bf16)
    cvt(x,          x_bf,   (size_t)MROWS * STIM);
    cvt(proj_w,     pw_bf,  (size_t)HID * STIM);
    cvt(in_proj_w,  ipw_bf, (size_t)2 * D_INNER * HID);
    cvt(x_proj_w,   xpw_bf, (size_t)NXDB * D_INNER);
    cvt(out_proj_w, opw_bf, (size_t)HID * D_INNER);
    cvt(head_w,     hw_bf,  (size_t)VOX * HID);

    const int GY = MROWS / TM;   // 128

    // 1) h = x @ proj_w.T + proj_b             -> bf16 (16384 x 256, K=1024)
    gemm_bf16_wmma<<<dim3(HID / TN, GY), 256, 0, stream>>>(
        x_bf, pw_bf, proj_b, h_bf, 1, MROWS, HID, STIM);

    // 2) xz = h @ in_proj_w.T                  -> f32 (16384 x 1024, K=256)
    gemm_bf16_wmma<<<dim3((2 * D_INNER) / TN, GY), 256, 0, stream>>>(
        h_bf, ipw_bf, nullptr, xz, 0, MROWS, 2 * D_INNER, HID);

    // 3) uc = silu(causal_conv(u))             -> f32 + bf16
    conv_silu_kernel<<<(MROWS * D_INNER) / 256, 256, 0, stream>>>(
        xz, conv_w, conv_b, uc, uc_bf);

    // 4) xdb = uc @ x_proj_w.T                 -> f32 (16384 x 48, K=512)
    gemm_bf16_wmma<<<dim3((NXDB + TN - 1) / TN, GY), 256, 0, stream>>>(
        uc_bf, xpw_bf, nullptr, xdb, 0, MROWS, NXDB, D_INNER);

    // 5) dt = softplus(dt_part @ dt_proj_w.T + b)
    dt_softplus_kernel<<<(MROWS * D_INNER) / 256, 256, 0, stream>>>(
        xdb, dt_proj_w, dt_proj_b, dtf);

    // 6) selective scan + D-skip + silu(z) gate -> bf16
    scan_kernel<<<NB, D_INNER, 0, stream>>>(
        dtf, xdb, uc, xz, A_log, Dp, yg_bf);

    // 7) o = yg @ out_proj_w.T                 -> bf16 (16384 x 256, K=512)
    gemm_bf16_wmma<<<dim3(HID / TN, GY), 256, 0, stream>>>(
        yg_bf, opw_bf, nullptr, o_bf, 1, MROWS, HID, D_INNER);

    // 8) out = o @ head_w.T + head_b           -> f32 (16384 x 8192, K=256)
    gemm_bf16_wmma<<<dim3(VOX / TN, GY), 256, 0, stream>>>(
        o_bf, hw_bf, head_b, out, 0, MROWS, VOX, HID);
}
